// CapsuleNet_12524124635980
// MI455X (gfx1250) — matrix operand order repaired and checked by reading, not compiled, not run on hardware
//
#include <hip/hip_runtime.h>
#include <hip/hip_bf16.h>
#include <math.h>
#include <stdint.h>

// ---------------------------------------------------------------------------
// CapsuleNet forward for MI455X (gfx1250, wave32, WMMA).
// Five GEMMs via v_wmma_f32_16x16x32_f16 with async global->LDS double
// buffering, + one wave-per-sample dynamic-routing kernel.
// ---------------------------------------------------------------------------

typedef __attribute__((ext_vector_type(16))) _Float16 v16h;
typedef __attribute__((ext_vector_type(8)))  _Float16 v8h;
typedef __attribute__((ext_vector_type(8)))  float    v8f;
typedef __attribute__((ext_vector_type(4)))  int      v4i_t;

union ABfrag { v16h v; v8h h[2]; };

#define CAPS_EPS 1e-8f

#if defined(__gfx1250__) && __has_builtin(__builtin_amdgcn_global_load_async_to_lds_b128)
#define USE_ASYNC_LDS 1
#else
#define USE_ASYNC_LDS 0
#endif

__device__ __forceinline__ void stage_pair(const _Float16* __restrict__ g,
                                           _Float16* __restrict__ s) {
#if USE_ASYNC_LDS
    // ROCm clang-22 signature: (int4* global_src, int4* lds_dst, imm offset, imm cpol)
    __builtin_amdgcn_global_load_async_to_lds_b128((v4i_t*)g,       (v4i_t*)s,       0, 0);
    __builtin_amdgcn_global_load_async_to_lds_b128((v4i_t*)(g + 8), (v4i_t*)(s + 8), 0, 0);
#else
    *(v8h*)s       = *(const v8h*)g;
    *(v8h*)(s + 8) = *(const v8h*)(g + 8);
#endif
}

__device__ __forceinline__ void wait_stage() {
#if USE_ASYNC_LDS
#if __has_builtin(__builtin_amdgcn_s_wait_asynccnt)
    __builtin_amdgcn_s_wait_asynccnt(0);
#else
    asm volatile("s_wait_asynccnt 0" ::: "memory");
#endif
#endif
    __syncthreads();
}

// ----------------------------- f32 -> f16 convert ---------------------------
__global__ void k_cvt_f16(const float* __restrict__ in, _Float16* __restrict__ out, int n) {
    int i = blockIdx.x * 256 + threadIdx.x;
    if (i < n) out[i] = (_Float16)in[i];
}

// ----------------------------- tiny utility kernels -------------------------
__global__ void k_zero1(float* p) { if (threadIdx.x == 0) p[0] = 0.f; }

__global__ void k_finalize_loss(const float* __restrict__ acc, float* __restrict__ out, float inv_n) {
    if (threadIdx.x == 0) out[0] = acc[0] * inv_n;
}

// ----------------------------- WMMA GEMM ------------------------------------
// C[M,N] = A[M,K] (f16, row-major) * B[N,K]^T (f16, weights stored [N][K]).
// Block tile 128(M) x 128(N), 256 threads = 8 waves (4M x 2N), each wave
// computes 32x64 = 2x4 WMMA tiles. Double-buffered LDS (async loads).
// EPI: 0 = store f32 (Cf), 1 = relu -> f16 (Ch), 2 = sigmoid -> f32 + MSE acc.
template<int EPI>
__global__ __launch_bounds__(256)
void k_gemm(const _Float16* __restrict__ A, const _Float16* __restrict__ B,
            const float* __restrict__ bias,
            float* __restrict__ Cf, _Float16* __restrict__ Ch,
            const float* __restrict__ ref, float* __restrict__ lossAcc,
            int M, int N, int K)
{
    // [buf][A: 128x32 | B: 128x32] halves -> 2 * 8192 * 2B = 32 KB
    __shared__ __align__(16) _Float16 smem[2 * 8192];

    const int t    = threadIdx.x;
    const int wid  = t >> 5;
    const int lane = t & 31;
    const int wm   = wid & 3;     // wave M position (0..3) -> 32 rows each
    const int wn   = wid >> 2;    // wave N position (0..1) -> 64 cols each
    const int lr   = lane & 15;
    const int lh   = lane >> 4;   // half-wave select

    const int rowBase = blockIdx.y * 128;
    const int colBase = blockIdx.x * 128;

    v8f acc[2][4];
#pragma unroll
    for (int m = 0; m < 2; ++m)
#pragma unroll
        for (int n = 0; n < 4; ++n)
#pragma unroll
            for (int r = 0; r < 8; ++r) acc[m][n][r] = 0.f;

    // staging coordinates: 256 threads cover a 128x32 f16 tile with 2 b128 each
    const int srow = t >> 1;          // 0..127
    const int scol = (t & 1) * 16;    // 0 / 16
    const int soff = srow * 32 + scol;
    const _Float16* gArow = A + (size_t)(rowBase + srow) * K + scol;
    const _Float16* gBrow = B + (size_t)(colBase + srow) * K + scol;

    // prologue: stage k0 = 0 into buffer 0
    stage_pair(gArow,         &smem[soff]);
    stage_pair(gBrow,         &smem[4096 + soff]);
    wait_stage();

    int buf = 0;
    for (int k0 = 0; k0 < K; k0 += 32) {
        if (k0 + 32 < K) {
            const int nb = (buf ^ 1) * 8192;
            stage_pair(gArow + (k0 + 32), &smem[nb + soff]);
            stage_pair(gBrow + (k0 + 32), &smem[nb + 4096 + soff]);
        }

        const _Float16* sAb = &smem[buf * 8192];
        const _Float16* sBb = sAb + 4096;

        ABfrag a[2], b[4];
#pragma unroll
        for (int m = 0; m < 2; ++m) {
            const int row = wm * 32 + m * 16 + lr;
            // ISA 16-bit A 16x32 layout: lanes 0-15 K{0..7,16..23}, lanes 16-31 K{8..15,24..31}
            a[m].h[0] = *(const v8h*)&sAb[row * 32 + (lh ? 8  : 0)];
            a[m].h[1] = *(const v8h*)&sAb[row * 32 + (lh ? 24 : 16)];
        }
#pragma unroll
        for (int n = 0; n < 4; ++n) {
            const int col = wn * 64 + n * 16 + lr;
            // ISA 16-bit B 32x16 layout: lanes 0-15 K{0..7,16..23}, lanes 16-31 K{8..15,24..31}
            b[n].h[0] = *(const v8h*)&sBb[col * 32 + (lh ? 8  : 0)];
            b[n].h[1] = *(const v8h*)&sBb[col * 32 + (lh ? 24 : 16)];
        }
#pragma unroll
        for (int m = 0; m < 2; ++m)
#pragma unroll
            for (int n = 0; n < 4; ++n)
                acc[m][n] = __builtin_amdgcn_wmma_f32_16x16x32_f16(
                    false, a[m].v, false, b[n].v, (short)0, acc[m][n], false, false);

        wait_stage();
        buf ^= 1;
    }

    float lsum = 0.f;
#pragma unroll
    for (int n = 0; n < 4; ++n) {
        const int gc = colBase + wn * 64 + n * 16 + lr;
        const float bv = bias ? bias[gc] : 0.f;
#pragma unroll
        for (int m = 0; m < 2; ++m) {
            const int rbase = rowBase + wm * 32 + m * 16 + lh * 8;
#pragma unroll
            for (int r = 0; r < 8; ++r) {
                const float v = acc[m][n][r] + bv;
                const size_t off = (size_t)(rbase + r) * N + gc;
                if (EPI == 0) {
                    Cf[off] = v;
                } else if (EPI == 1) {
                    Ch[off] = (_Float16)fmaxf(v, 0.f);
                } else {
                    const float s = 1.f / (1.f + __expf(-v));
                    Cf[off] = s;
                    const float d = s - ref[off];
                    lsum += d * d;
                }
            }
        }
    }
    if (EPI == 2) {
#pragma unroll
        for (int o = 16; o >= 1; o >>= 1) lsum += __shfl_xor(lsum, o);
        if (lane == 0) atomicAdd(lossAcc, lsum);
    }
}

// ----------------------- squash + x_hat + dynamic routing -------------------
// One wave per batch sample; lane = input capsule (32). 8 samples / block.
__global__ __launch_bounds__(256)
void k_caps_route(const float* __restrict__ P,    // [B,256] pconv output (f32)
                  const float* __restrict__ W,    // caps_w [10,32,16,8] f32
                  float* __restrict__ outLen,     // [B,10]
                  _Float16* __restrict__ masked)  // [B,160] f16 (one-hot masked v)
{
    __shared__ float vsh[8][160];
    const int wid  = threadIdx.x >> 5;
    const int lane = threadIdx.x & 31;
    const int b    = blockIdx.x * 8 + wid;

    // u = squash(p[b, lane*8 .. lane*8+7])
    float u[8];
    {
        const float* p = P + (size_t)b * 256 + lane * 8;
        float n2 = 0.f;
#pragma unroll
        for (int c = 0; c < 8; ++c) { u[c] = p[c]; n2 += u[c] * u[c]; }
        const float sc = n2 / (1.f + n2) / (sqrtf(n2) + CAPS_EPS);
#pragma unroll
        for (int c = 0; c < 8; ++c) u[c] *= sc;
    }

    // x_hat[o][d] for this lane's input capsule (kept in registers)
    float xh[10][16];
#pragma unroll
    for (int o = 0; o < 10; ++o) {
        const float* w = W + ((size_t)o * 32 + lane) * 128;  // 16x8 slab
#pragma unroll
        for (int d = 0; d < 16; ++d) {
            float s = 0.f;
#pragma unroll
            for (int c = 0; c < 8; ++c) s += w[d * 8 + c] * u[c];
            xh[o][d] = s;
        }
    }

    float bl[10];
#pragma unroll
    for (int o = 0; o < 10; ++o) bl[o] = 0.f;
    float len[10];

#pragma unroll
    for (int it = 0; it < 3; ++it) {
        // softmax over output capsules (lane-local)
        float mx = bl[0];
#pragma unroll
        for (int o = 1; o < 10; ++o) mx = fmaxf(mx, bl[o]);
        float cc[10]; float ssum = 0.f;
#pragma unroll
        for (int o = 0; o < 10; ++o) { cc[o] = __expf(bl[o] - mx); ssum += cc[o]; }
        const float inv = 1.f / ssum;
#pragma unroll
        for (int o = 0; o < 10; ++o) cc[o] *= inv;

#pragma unroll
        for (int o = 0; o < 10; ++o) {
            float s[16];
#pragma unroll
            for (int d = 0; d < 16; ++d) {
                float v = cc[o] * xh[o][d];
#pragma unroll
                for (int off = 16; off >= 1; off >>= 1) v += __shfl_xor(v, off);
                s[d] = v;   // full sum over i, present in all lanes
            }
            float n2 = 0.f;
#pragma unroll
            for (int d = 0; d < 16; ++d) n2 += s[d] * s[d];
            const float sc = n2 / (1.f + n2) / (sqrtf(n2) + CAPS_EPS);
            if (it < 2) {
                float dot = 0.f;
#pragma unroll
                for (int d = 0; d < 16; ++d) dot += sc * s[d] * xh[o][d];
                bl[o] += dot;
            } else {
                len[o] = sc * sqrtf(n2);
                if (lane == 0) {
#pragma unroll
                    for (int d = 0; d < 16; ++d) vsh[wid][o * 16 + d] = sc * s[d];
                }
            }
        }
    }

    if (lane == 0) {
#pragma unroll
        for (int o = 0; o < 10; ++o) outLen[(size_t)b * 10 + o] = len[o];
    }
    // argmax (first max, matching jnp.argmax)
    int best = 0; float bestv = len[0];
#pragma unroll
    for (int o = 1; o < 10; ++o) if (len[o] > bestv) { bestv = len[o]; best = o; }

    __syncthreads();
#pragma unroll
    for (int j = 0; j < 5; ++j) {
        const int idx = lane * 5 + j;
        const int o = idx >> 4;
        const float v = (o == best) ? vsh[wid][idx] : 0.f;
        masked[(size_t)b * 160 + idx] = (_Float16)v;
    }
}

// ----------------------------- host launcher --------------------------------
static inline char* carve(char*& p, size_t bytes) {
    char* r = p;
    p += (bytes + 255) & ~(size_t)255;
    return r;
}

extern "C" void kernel_launch(void* const* d_in, const int* in_sizes, int n_in,
                              void* d_out, int out_size, void* d_ws, size_t ws_size,
                              hipStream_t stream) {
    (void)in_sizes; (void)n_in; (void)out_size; (void)ws_size;

    const int B = 8192, O = 10;
    const float* x       = (const float*)d_in[0];
    const float* x1      = (const float*)d_in[1];
    const float* conv1_w = (const float*)d_in[2];
    const float* conv1_b = (const float*)d_in[3];
    const float* pconv_w = (const float*)d_in[4];
    const float* pconv_b = (const float*)d_in[5];
    const float* caps_w  = (const float*)d_in[6];
    const float* dec_w1  = (const float*)d_in[7];
    const float* dec_b1  = (const float*)d_in[8];
    const float* dec_w2  = (const float*)d_in[9];
    const float* dec_b2  = (const float*)d_in[10];
    const float* dec_w3  = (const float*)d_in[11];
    const float* dec_b3  = (const float*)d_in[12];

    float* out      = (float*)d_out;
    float* outLen   = out;                         // [B,10]
    float* outRecon = out + (size_t)B * O;         // [B,1024]
    float* outLoss  = outRecon + (size_t)B * 1024; // scalar

    char* p = (char*)d_ws;
    _Float16* xh  = (_Float16*)carve(p, (size_t)B * 1024 * 2);  // x as f16
    _Float16* w1h = (_Float16*)carve(p, (size_t)256 * 1024 * 2);
    _Float16* h1h = (_Float16*)carve(p, (size_t)B * 256 * 2);   // conv1 relu out
    _Float16* w2h = (_Float16*)carve(p, (size_t)256 * 256 * 2);
    float*    pf  = (float*)   carve(p, (size_t)B * 256 * 4);   // pconv out f32
    _Float16* w3h = (_Float16*)carve(p, (size_t)512 * 160 * 2);
    _Float16* a1h = (_Float16*)carve(p, (size_t)B * 160 * 2);   // masked f16
    _Float16* d1h = (_Float16*)carve(p, (size_t)B * 512 * 2);
    _Float16* w4h = (_Float16*)carve(p, (size_t)1024 * 512 * 2);
    _Float16* d2h = (_Float16*)carve(p, (size_t)B * 1024 * 2);
    _Float16* w5h = (_Float16*)carve(p, (size_t)1024 * 1024 * 2);
    float*    acc = (float*)   carve(p, 256);

    auto cvt = [&](const float* src, _Float16* dst, int n) {
        k_cvt_f16<<<(n + 255) / 256, 256, 0, stream>>>(src, dst, n);
    };
    cvt(x,       xh,  B * 1024);
    cvt(conv1_w, w1h, 256 * 1024);
    cvt(pconv_w, w2h, 256 * 256);
    cvt(dec_w1,  w3h, 512 * 160);
    cvt(dec_w2,  w4h, 1024 * 512);
    cvt(dec_w3,  w5h, 1024 * 1024);

    const dim3 blk(256);
    // conv1: [B,1024] x [256,1024]^T -> relu f16 [B,256]
    k_gemm<1><<<dim3(256 / 128, B / 128), blk, 0, stream>>>(
        xh, w1h, conv1_b, nullptr, h1h, nullptr, nullptr, B, 256, 1024);
    // pconv: [B,256] x [256,256]^T -> f32 [B,256]
    k_gemm<0><<<dim3(256 / 128, B / 128), blk, 0, stream>>>(
        h1h, w2h, pconv_b, pf, nullptr, nullptr, nullptr, B, 256, 256);
    // squash + x_hat + routing + mask
    k_caps_route<<<B / 8, blk, 0, stream>>>(pf, caps_w, outLen, a1h);
    // decoder
    k_gemm<1><<<dim3(512 / 128, B / 128), blk, 0, stream>>>(
        a1h, w3h, dec_b1, nullptr, d1h, nullptr, nullptr, B, 512, 160);
    k_gemm<1><<<dim3(1024 / 128, B / 128), blk, 0, stream>>>(
        d1h, w4h, dec_b2, nullptr, d2h, nullptr, nullptr, B, 1024, 512);
    k_zero1<<<1, 32, 0, stream>>>(acc);
    k_gemm<2><<<dim3(1024 / 128, B / 128), blk, 0, stream>>>(
        d2h, w5h, dec_b3, outRecon, nullptr, x1, acc, B, 1024, 1024);
    k_finalize_loss<<<1, 32, 0, stream>>>(acc, outLoss, 1.f / (float)((size_t)B * 1024));
}
